// LSTM_15118284882300
// MI455X (gfx1250) — matrix operand order repaired and checked by reading, not compiled, not run on hardware
//
#include <hip/hip_runtime.h>

// 3-layer fused LSTM (H=32, B=512, T=2048) + FC, single persistent kernel.
// One workgroup = 16 batch rows, 8 waves; wave w owns gate columns [16w,16w+16).
// Per step: layer0 = 1 WMMA/wave, layers 1,2 = 2 WMMAs/wave (K=32 == one
// v_wmma_f32_16x16x32_f16). h state in LDS (f16, padded rows), c in registers.
// v3: x[t] loaded via wave-uniform scalar loads (KMcnt — not drained by the
// barrier's loadcnt/dscnt wait, so the 1-step prefetch truly floats), and
// activations use native v_tanh_f32 where available.

typedef __attribute__((ext_vector_type(16))) _Float16 v16h;
typedef __attribute__((ext_vector_type(8)))  float    v8f;
typedef __attribute__((ext_vector_type(4)))  float    v4f;

#define HID   32
#define SEQT  2048
#define GSTR  132   // padded gates row stride (floats) -> conflict-light
#define HSTR  40    // padded h row stride (halves) = 80B, 16B-aligned rows

#define WMMA_F16(A, Bm, C) \
  __builtin_amdgcn_wmma_f32_16x16x32_f16(false, (A), false, (Bm), (short)0, (C), false, false)

#if __has_builtin(__builtin_amdgcn_tanhf)
__device__ __forceinline__ float fast_tanh(float x) {
    return __builtin_amdgcn_tanhf(x);            // v_tanh_f32 (gfx1250 TRANS op)
}
__device__ __forceinline__ float fast_sigmoid(float x) {
    return __builtin_fmaf(0.5f, __builtin_amdgcn_tanhf(0.5f * x), 0.5f);
}
#else
__device__ __forceinline__ float fast_tanh(float x) {
    return 1.f - 2.f * __builtin_amdgcn_rcpf(1.f + __expf(2.f * x));
}
__device__ __forceinline__ float fast_sigmoid(float x) {
    return __builtin_amdgcn_rcpf(1.f + __expf(-x));
}
#endif

__global__ __launch_bounds__(256, 1)
void lstm3_fused(const float* __restrict__ x,
                 const float* __restrict__ wih0, const float* __restrict__ whh0,
                 const float* __restrict__ bih0, const float* __restrict__ bhh0,
                 const float* __restrict__ wih1, const float* __restrict__ whh1,
                 const float* __restrict__ bih1, const float* __restrict__ bhh1,
                 const float* __restrict__ wih2, const float* __restrict__ whh2,
                 const float* __restrict__ bih2, const float* __restrict__ bhh2,
                 const float* __restrict__ fcw,  const float* __restrict__ fcb,
                 float* __restrict__ out)
{
    __shared__ __align__(16) _Float16 hbuf[3][16][HSTR]; // h state per layer, f16
    __shared__ float gates[16][GSTR];                    // gate pre-activations

    const int tid  = threadIdx.x;
    const int lane = tid & 31;
    const int wave = tid >> 5;        // N-tile id 0..7
    const int nl   = lane & 15;
    const int half = lane >> 4;

    // ---- B-matrix weights: row = global N, K-half by lane-half; f32 -> f16
    const int brow = wave * 16 + nl;
    const int bkb  = half * 16;
    auto loadB = [&](const float* w) -> v16h {
        v16h r;
        const float* p = w + brow * HID + bkb;
        #pragma unroll
        for (int j = 0; j < 16; ++j) r[j] = (_Float16)p[j];
        return r;
    };
    const v16h Bwh0 = loadB(whh0);
    const v16h Bwi1 = loadB(wih1);
    const v16h Bwh1 = loadB(whh1);
    const v16h Bwi2 = loadB(wih2);
    const v16h Bwh2 = loadB(whh2);

    // ---- A-matrix lane coords (16-bit A 16x32 layout)
    const int am  = lane & 15;        // M row
    const int ak0 = half * 8;         // K chunk0 start = 2*ak0 (0 or 16); chunk1 = +8

    // ---- elementwise coords: thread -> (batch row, 2 h indices)
    const int eb = tid >> 4;          // 0..15  (== 2*wave + half)
    const int eh = (tid & 15) * 2;    // h base; handles eh, eh+1

    // biases (b_ih + b_hh) per thread, per layer, per gate
    float bi[3][2], bf[3][2], bg[3][2], bo[3][2];
    {
        const float* bihs[3] = {bih0, bih1, bih2};
        const float* bhhs[3] = {bhh0, bhh1, bhh2};
        #pragma unroll
        for (int l = 0; l < 3; ++l) {
            #pragma unroll
            for (int j = 0; j < 2; ++j) {
                const int h = eh + j;
                bi[l][j] = bihs[l][h]      + bhhs[l][h];
                bf[l][j] = bihs[l][h + 32] + bhhs[l][h + 32];
                bg[l][j] = bihs[l][h + 64] + bhhs[l][h + 64];
                bo[l][j] = bihs[l][h + 96] + bhhs[l][h + 96];
            }
        }
    }
    // layer-0 input weights (in_dim == 1): column vector per gate
    float w0i[2], w0f[2], w0g[2], w0o[2];
    #pragma unroll
    for (int j = 0; j < 2; ++j) {
        w0i[j] = wih0[eh + j];
        w0f[j] = wih0[32 + eh + j];
        w0g[j] = wih0[64 + eh + j];
        w0o[j] = wih0[96 + eh + j];
    }

    float creg[3][2] = {{0.f, 0.f}, {0.f, 0.f}, {0.f, 0.f}};

    // zero the h state (only the live 32 columns are ever read)
    for (int idx = tid; idx < 3 * 16 * HID; idx += 256) {
        const int l = idx / (16 * HID);
        const int r = (idx / HID) & 15;
        const int k = idx % HID;
        hbuf[l][r][k] = (_Float16)0.f;
    }
    __syncthreads();

    // ---- wave-uniform scalar pointers to the wave's two x rows (rows 2w, 2w+1)
    const int wrow = __builtin_amdgcn_readfirstlane(wave * 2);
    const float* xr0 = x + (blockIdx.x * 16 + wrow) * SEQT;  // uniform -> s_load
    const float* xr1 = xr0 + SEQT;

    auto loadA = [&](int l) -> v16h {
        union { v16h h; v4f f[2]; } u;
        const v4f* p = (const v4f*)(&hbuf[l][am][0]);   // row is 16B-aligned (80B stride)
        u.f[0] = p[2 * (ak0 >> 3)];      // K = 2*ak0 .. 2*ak0+7
        u.f[1] = p[2 * (ak0 >> 3) + 1];  // K = 2*ak0+8 .. 2*ak0+15
        return u.h;
    };

    auto storeGates = [&](v8f acc) {
        const int col = wave * 16 + nl;
        const int m0  = half * 8;
        #pragma unroll
        for (int v = 0; v < 8; ++v) gates[m0 + v][col] = acc[v];
    };

    auto cell = [&](int l, float xv) {
        #pragma unroll
        for (int j = 0; j < 2; ++j) {
            const int h = eh + j;
            float gi = gates[eb][h]      + bi[l][j];
            float gf = gates[eb][h + 32] + bf[l][j];
            float gg = gates[eb][h + 64] + bg[l][j];
            float go = gates[eb][h + 96] + bo[l][j];
            if (l == 0) {
                gi += xv * w0i[j]; gf += xv * w0f[j];
                gg += xv * w0g[j]; go += xv * w0o[j];
            }
            gi = fast_sigmoid(gi);
            gf = fast_sigmoid(gf);
            gg = fast_tanh(gg);
            go = fast_sigmoid(go);
            const float c = gf * creg[l][j] + gi * gg;
            creg[l][j] = c;
            const float hn = go * fast_tanh(c);
            hbuf[l][eb][h] = (_Float16)hn;
        }
    };

    // pipelined scalar x values for the current step
    float xs0 = xr0[0];
    float xs1 = xr1[0];
    for (int t = 0; t < SEQT; ++t) {
        // issue next step's scalar x loads early (KMcnt: not drained by barriers)
        const int tn = (t + 1 < SEQT) ? (t + 1) : t;
        const float xn0 = xr0[tn];
        const float xn1 = xr1[tn];

        // ---- layer 0: gates = h0 @ whh0^T   (x*wih0 folded into cell)
        v8f acc = {0.f, 0.f, 0.f, 0.f, 0.f, 0.f, 0.f, 0.f};
        acc = WMMA_F16(loadA(0), Bwh0, acc);
        storeGates(acc);
        __syncthreads();
        cell(0, half ? xs1 : xs0);
        __syncthreads();

        // ---- layer 1: gates = h0 @ wih1^T + h1 @ whh1^T
        acc = (v8f){0.f, 0.f, 0.f, 0.f, 0.f, 0.f, 0.f, 0.f};
        acc = WMMA_F16(loadA(0), Bwi1, acc);
        acc = WMMA_F16(loadA(1), Bwh1, acc);
        storeGates(acc);
        __syncthreads();
        cell(1, 0.f);
        __syncthreads();

        // ---- layer 2: gates = h1 @ wih2^T + h2 @ whh2^T
        acc = (v8f){0.f, 0.f, 0.f, 0.f, 0.f, 0.f, 0.f, 0.f};
        acc = WMMA_F16(loadA(1), Bwi2, acc);
        acc = WMMA_F16(loadA(2), Bwh2, acc);
        storeGates(acc);
        __syncthreads();
        cell(2, 0.f);
        __syncthreads();

        xs0 = xn0;
        xs1 = xn1;
    }

    // ---- FC on last timestep of layer 2: out[b] = h2[b,:] . fcw + fcb
    if (tid < 16) {
        float s = fcb[0];
        #pragma unroll
        for (int h = 0; h < HID; ++h)
            s += (float)hbuf[2][tid][h] * fcw[h];
        out[blockIdx.x * 16 + tid] = s;
    }
}

extern "C" void kernel_launch(void* const* d_in, const int* in_sizes, int n_in,
                              void* d_out, int out_size, void* d_ws, size_t ws_size,
                              hipStream_t stream) {
    (void)in_sizes; (void)n_in; (void)d_ws; (void)ws_size; (void)out_size;
    const float* x    = (const float*)d_in[0];
    const float* wih0 = (const float*)d_in[1];
    const float* whh0 = (const float*)d_in[2];
    const float* bih0 = (const float*)d_in[3];
    const float* bhh0 = (const float*)d_in[4];
    const float* wih1 = (const float*)d_in[5];
    const float* whh1 = (const float*)d_in[6];
    const float* bih1 = (const float*)d_in[7];
    const float* bhh1 = (const float*)d_in[8];
    const float* wih2 = (const float*)d_in[9];
    const float* whh2 = (const float*)d_in[10];
    const float* bih2 = (const float*)d_in[11];
    const float* bhh2 = (const float*)d_in[12];
    const float* fcw  = (const float*)d_in[13];
    const float* fcb  = (const float*)d_in[14];
    float* out = (float*)d_out;

    // 512 batch rows / 16 per workgroup = 32 workgroups, 256 threads (8 waves)
    lstm3_fused<<<dim3(32), dim3(256), 0, stream>>>(
        x, wih0, whh0, bih0, bhh0,
        wih1, whh1, bih1, bhh1,
        wih2, whh2, bih2, bhh2,
        fcw, fcb, out);
}
